// GCN_MLP_53412213293178
// MI455X (gfx1250) — matrix operand order repaired and checked by reading, not compiled, not run on hardware
//
#include <hip/hip_runtime.h>
#include <hip/hip_bf16.h>

// ---------------------------------------------------------------------------
// GCN (6x GCNConv + LN/ReLU + residual + mean-pool + MLP head) for MI455X.
// GEMMs: v_wmma_f32_16x16x32_f16 with the weight matrix staged into LDS via
// gfx1250 GLOBAL_LOAD_ASYNC_TO_LDS_B128 (ASYNCcnt) and read back one half per LDS load.
// Edge aggregation uses native f32 global atomics (bandwidth-bound path).
// ---------------------------------------------------------------------------

typedef __attribute__((ext_vector_type(16))) _Float16 v16h;
typedef __attribute__((ext_vector_type(8)))  float    v8f;

#define NN    100000
#define NE    1600000
#define FIN   128
#define HD    64
#define NG    64
#define LNEPS 1e-5f

// ---------------------------- utility kernels ------------------------------

__global__ void k_zero(float* __restrict__ p, int n) {
  int i = blockIdx.x * blockDim.x + threadIdx.x;
  if (i < n) p[i] = 0.0f;
}

__global__ void k_cvt_h(const float* __restrict__ w, _Float16* __restrict__ wh, int n) {
  int i = blockIdx.x * blockDim.x + threadIdx.x;
  if (i < n) wh[i] = (_Float16)w[i];
}

// degree (scatter count of dst)
__global__ void k_deg(const int* __restrict__ dst, float* __restrict__ deg, int e) {
  int i = blockIdx.x * blockDim.x + threadIdx.x;
  if (i < e) atomicAdd(&deg[dst[i]], 1.0f);
}

// dis = rsqrt(deg+1), dis2 = 1/(deg+1)
__global__ void k_dis(const float* __restrict__ deg, float* __restrict__ dis,
                      float* __restrict__ dis2, int n) {
  int i = blockIdx.x * blockDim.x + threadIdx.x;
  if (i < n) {
    float d = deg[i] + 1.0f;
    float r = rsqrtf(d);
    dis[i]  = r;
    dis2[i] = 1.0f / d;
  }
}

// per-edge norm = dis[src]*dis[dst]  (same for every layer -> compute once)
__global__ void k_enorm(const int* __restrict__ src, const int* __restrict__ dst,
                        const float* __restrict__ dis, float* __restrict__ en, int e) {
  int i = blockIdx.x * blockDim.x + threadIdx.x;
  if (i < e) en[i] = dis[src[i]] * dis[dst[i]];
}

// ------------------------------- WMMA GEMM ---------------------------------
// C[M x 64] = A[M x K] (f32, row major) * Bh[K x 64] (f16, row major)
// One wave -> one 16(M) x 64(N) slab; K stepped by 32.
// Bh (<=16KB) is staged block-wide into LDS with async-to-LDS, then each
// wave's B fragments are gathered one half at a time from LDS.
__global__ void k_gemm_wmma(const float* __restrict__ A, const _Float16* __restrict__ Bh,
                            float* __restrict__ C, int M, int K) {
  __shared__ _Float16 lw[FIN * HD];          // 16 KB max (K*HD used)

  // ---- stage weights: global -> LDS, 16B per lane-op, ASYNCcnt tracked ----
  {
    int nchunk = (K * HD * (int)sizeof(_Float16)) >> 4;   // # of 16B chunks
    const char* gbase = (const char*)Bh;
    unsigned lbase = (unsigned)(uintptr_t)(&lw[0]);       // LDS addr = addr[31:0]
    for (int c = threadIdx.x; c < nchunk; c += blockDim.x) {
      unsigned laddr = lbase + ((unsigned)c << 4);
      const char* gaddr = gbase + ((size_t)c << 4);
      asm volatile("global_load_async_to_lds_b128 %0, %1, off"
                   :: "v"(laddr), "v"(gaddr) : "memory");
    }
    asm volatile("s_wait_asynccnt 0x0" ::: "memory");
  }
  __syncthreads();

  int gtid = blockIdx.x * blockDim.x + threadIdx.x;
  int wave = gtid >> 5;
  int lane = threadIdx.x & 31;
  int m0 = wave * 16;
  if (m0 >= M) return;                 // wave-uniform: EXEC all-1s inside WMMA

  int row = lane & 15;                 // A: M index for this lane
  int hi  = lane >> 4;                 // A: which K-half group
  const float* arow = A + (size_t)(m0 + row) * K;

  v8f acc0 = {}, acc1 = {}, acc2 = {}, acc3 = {};

  for (int k0 = 0; k0 < K; k0 += 32) {
    // --- A fragment (16-bit A 16x32 layout, ISA 7.12.2) ---
    v16h a;
#pragma unroll
    for (int v = 0; v < 8; ++v) {
      int group = v >> 2, vv = v & 3;
      int k = k0 + group * 16 + hi * 8 + 2 * vv;
      float2 f2 = *(const float2*)(arow + k);
      a[2 * v]     = (_Float16)f2.x;
      a[2 * v + 1] = (_Float16)f2.y;
    }
    // --- B fragments from LDS: lane holds B column N=lane%16, 16 K halfs (same K order as A) ---
    v16h b0, b1, b2, b3;
#pragma unroll
    for (int s = 0; s < 16; ++s) {
      int kk = (s < 8) ? (hi * 8 + s) : (16 + hi * 8 + (s - 8));
      const _Float16* brow = &lw[(k0 + kk) * HD + (lane & 15)];
      b0[s] = brow[0];
      b1[s] = brow[16];
      b2[s] = brow[32];
      b3[s] = brow[48];
    }

    acc0 = __builtin_amdgcn_wmma_f32_16x16x32_f16(false, a, false, b0, (short)0, acc0, false, false);
    acc1 = __builtin_amdgcn_wmma_f32_16x16x32_f16(false, a, false, b1, (short)0, acc1, false, false);
    acc2 = __builtin_amdgcn_wmma_f32_16x16x32_f16(false, a, false, b2, (short)0, acc2, false, false);
    acc3 = __builtin_amdgcn_wmma_f32_16x16x32_f16(false, a, false, b3, (short)0, acc3, false, false);
  }

  // --- store: 32-bit C/D 16x16 layout: N = lane%16, M = hi*8 + r ---
  int ncol = lane & 15;
  float* cbase = C + (size_t)m0 * HD + ncol;
#pragma unroll
  for (int r = 0; r < 8; ++r) {
    size_t mo = (size_t)(hi * 8 + r) * HD;
    cbase[mo +  0] = acc0[r];
    cbase[mo + 16] = acc1[r];
    cbase[mo + 32] = acc2[r];
    cbase[mo + 48] = acc3[r];
  }
}

// --------------------------- GCN message passing ---------------------------

// out[i,f] = g[i,f]*dis2[i] + b[f]   (self-loop term + bias; also initializes out)
__global__ void k_selfloop(const float* __restrict__ g, const float* __restrict__ dis2,
                           const float* __restrict__ b, float* __restrict__ out, int n) {
  long long idx = (long long)blockIdx.x * blockDim.x + threadIdx.x;
  if (idx >= (long long)n * HD) return;
  int node = (int)(idx >> 6);
  int f = (int)idx & 63;
  out[idx] = g[idx] * dis2[node] + b[f];
}

// scatter: out[dst] += g[src] * enorm  (edge uniform per wave, 2 feats/lane)
__global__ void k_scatter(const int* __restrict__ src, const int* __restrict__ dst,
                          const float* __restrict__ en, const float* __restrict__ g,
                          float* __restrict__ out, int e) {
  long long idx = (long long)blockIdx.x * blockDim.x + threadIdx.x;
  long long ee = idx >> 5;
  if (ee >= e) return;
  int f = ((int)idx & 31) * 2;
  int s = src[ee], d = dst[ee];
  float nrm = en[ee];
  float2 gv = *(const float2*)(g + (size_t)s * HD + f);
  float* op = out + (size_t)d * HD + f;
  atomicAdd(op,     gv.x * nrm);
  atomicAdd(op + 1, gv.y * nrm);
}

// fused LayerNorm(64) + ReLU, wave per node (lane handles f and f+32)
__global__ void k_ln_relu(float* __restrict__ h, const float* __restrict__ gam,
                          const float* __restrict__ bet, int n) {
  int wave = (blockIdx.x * blockDim.x + threadIdx.x) >> 5;
  int lane = threadIdx.x & 31;
  if (wave >= n) return;
  float* row = h + (size_t)wave * HD;
  float x0 = row[lane], x1 = row[lane + 32];
  float s = x0 + x1;
#pragma unroll
  for (int m = 16; m >= 1; m >>= 1) s += __shfl_xor(s, m, 32);
  float mu = s * (1.0f / 64.0f);
  float d0 = x0 - mu, d1 = x1 - mu;
  float v = d0 * d0 + d1 * d1;
#pragma unroll
  for (int m = 16; m >= 1; m >>= 1) v += __shfl_xor(v, m, 32);
  float inv = rsqrtf(v * (1.0f / 64.0f) + LNEPS);
  float y0 = d0 * inv * gam[lane]      + bet[lane];
  float y1 = d1 * inv * gam[lane + 32] + bet[lane + 32];
  row[lane]      = fmaxf(y0, 0.0f);
  row[lane + 32] = fmaxf(y1, 0.0f);
}

// h += r + bres[f]
__global__ void k_res_add(float* __restrict__ h, const float* __restrict__ r,
                          const float* __restrict__ bres, int n) {
  long long idx = (long long)blockIdx.x * blockDim.x + threadIdx.x;
  if (idx >= (long long)n * HD) return;
  h[idx] += r[idx] + bres[(int)idx & 63];
}

// ------------------------------- pooling / head ----------------------------

__global__ void k_pool(const float* __restrict__ h, const int* __restrict__ batch,
                       float* __restrict__ sums, float* __restrict__ cnt, int n) {
  int wave = (blockIdx.x * blockDim.x + threadIdx.x) >> 5;
  int lane = threadIdx.x & 31;
  if (wave >= n) return;
  int gid = batch[wave];
  const float* row = h + (size_t)wave * HD;
  atomicAdd(&sums[(size_t)gid * HD + lane],      row[lane]);
  atomicAdd(&sums[(size_t)gid * HD + lane + 32], row[lane + 32]);
  if (lane == 0) atomicAdd(&cnt[gid], 1.0f);
}

// out[g] = relu(pooled @ Wm1 + bm1) @ Wm2 + bm2    (64 graphs, 1 thread each)
__global__ void k_mlp(const float* __restrict__ sums, const float* __restrict__ cnt,
                      const float* __restrict__ Wm1, const float* __restrict__ bm1,
                      const float* __restrict__ Wm2, const float* __restrict__ bm2,
                      float* __restrict__ out) {
  int g = threadIdx.x;
  if (g >= NG) return;
  float c = fmaxf(cnt[g], 1.0f);
  float p[HD];
#pragma unroll 8
  for (int k = 0; k < HD; ++k) p[k] = sums[(size_t)g * HD + k] / c;
  float o = bm2[0];
  for (int j = 0; j < HD; ++j) {
    float t = bm1[j];
#pragma unroll 8
    for (int k = 0; k < HD; ++k) t += p[k] * Wm1[(size_t)k * HD + j];
    o += fmaxf(t, 0.0f) * Wm2[j];
  }
  out[g] = o;
}

// ------------------------------- host driver -------------------------------

extern "C" void kernel_launch(void* const* d_in, const int* in_sizes, int n_in,
                              void* d_out, int out_size, void* d_ws, size_t ws_size,
                              hipStream_t stream) {
  (void)in_sizes; (void)n_in; (void)out_size; (void)ws_size;

  const float* x     = (const float*)d_in[0];
  const int*   ei    = (const int*)d_in[1];
  const int*   src   = ei;
  const int*   dst   = ei + NE;
  const int*   batch = (const int*)d_in[2];
  const float* W[6], *b[6];
  for (int l = 0; l < 6; ++l) { W[l] = (const float*)d_in[3 + 2 * l]; b[l] = (const float*)d_in[4 + 2 * l]; }
  const float* gam[5], *bet[5];
  for (int l = 0; l < 5; ++l) { gam[l] = (const float*)d_in[15 + 2 * l]; bet[l] = (const float*)d_in[16 + 2 * l]; }
  const float* Wres = (const float*)d_in[25];
  const float* bres = (const float*)d_in[26];
  const float* Wm1  = (const float*)d_in[27];
  const float* bm1  = (const float*)d_in[28];
  const float* Wm2  = (const float*)d_in[29];
  const float* bm2  = (const float*)d_in[30];
  float* out = (float*)d_out;

  // ---- carve workspace (256B aligned slabs) ----
  char* base = (char*)d_ws;
  size_t off = 0;
  auto carve = [&](size_t bytes) -> void* {
    off = (off + 255) & ~(size_t)255;
    void* p = base + off;
    off += bytes;
    return p;
  };
  const size_t nodeBytes = (size_t)NN * HD * sizeof(float);
  float* bufA  = (float*)carve(nodeBytes);   // gemm output / residual scratch
  float* bufB  = (float*)carve(nodeBytes);   // activations (ping)
  float* bufC  = (float*)carve(nodeBytes);   // activations (pong)
  float* deg   = (float*)carve((size_t)NN * sizeof(float));
  float* dis   = (float*)carve((size_t)NN * sizeof(float));
  float* dis2  = (float*)carve((size_t)NN * sizeof(float));
  float* en    = (float*)carve((size_t)NE * sizeof(float));
  _Float16* W1h   = (_Float16*)carve((size_t)FIN * HD * sizeof(_Float16));
  _Float16* Wh[5];
  for (int l = 0; l < 5; ++l) Wh[l] = (_Float16*)carve((size_t)HD * HD * sizeof(_Float16));
  _Float16* Wresh = (_Float16*)carve((size_t)HD * HD * sizeof(_Float16));
  float* pooled = (float*)carve((size_t)NG * HD * sizeof(float));
  float* cnt    = (float*)carve((size_t)NG * sizeof(float));

  const int B256 = 256;
  auto gridFor = [](long long threads) { return (int)((threads + 255) / 256); };

  // ---- degrees / norms (layer-invariant) ----
  k_zero<<<gridFor(NN), B256, 0, stream>>>(deg, NN);
  k_deg<<<gridFor(NE), B256, 0, stream>>>(dst, deg, NE);
  k_dis<<<gridFor(NN), B256, 0, stream>>>(deg, dis, dis2, NN);
  k_enorm<<<gridFor(NE), B256, 0, stream>>>(src, dst, dis, en, NE);

  // ---- convert weights to f16 for WMMA B operands ----
  k_cvt_h<<<gridFor(FIN * HD), B256, 0, stream>>>(W[0], W1h, FIN * HD);
  for (int l = 0; l < 5; ++l)
    k_cvt_h<<<gridFor(HD * HD), B256, 0, stream>>>(W[l + 1], Wh[l], HD * HD);
  k_cvt_h<<<gridFor(HD * HD), B256, 0, stream>>>(Wres, Wresh, HD * HD);

  const long long gemmThreads = (long long)((NN + 15) / 16) * 32;
  const long long elemThreads = (long long)NN * HD;
  const long long waveThreads = (long long)NN * 32;
  const long long edgeThreads = (long long)NE * 32;

  // conv(in, Whalf, bias, outBuf): gemm -> selfloop init -> edge scatter
  auto conv = [&](const float* in, const _Float16* wh, const float* bias,
                  float* outBuf, int K) {
    k_gemm_wmma<<<gridFor(gemmThreads), B256, 0, stream>>>(in, wh, bufA, NN, K);
    k_selfloop<<<gridFor(elemThreads), B256, 0, stream>>>(bufA, dis2, bias, outBuf, NN);
    k_scatter<<<gridFor(edgeThreads), B256, 0, stream>>>(src, dst, en, bufA, outBuf, NE);
  };

  // L1: x -> bufB ; bufB doubles as x_res (preserved until after L3)
  conv(x, W1h, b[0], bufB, FIN);
  k_ln_relu<<<gridFor(waveThreads), B256, 0, stream>>>(bufB, gam[0], bet[0], NN);

  // L2: bufB -> bufC
  conv(bufB, Wh[0], b[1], bufC, HD);
  k_ln_relu<<<gridFor(waveThreads), B256, 0, stream>>>(bufC, gam[1], bet[1], NN);

  // L3: bufC -> bufC (bufC source consumed by gemm before overwrite)
  conv(bufC, Wh[1], b[2], bufC, HD);
  k_ln_relu<<<gridFor(waveThreads), B256, 0, stream>>>(bufC, gam[2], bet[2], NN);

  // residual: bufC += x_res @ Wres + bres
  k_gemm_wmma<<<gridFor(gemmThreads), B256, 0, stream>>>(bufB, Wresh, bufA, NN, HD);
  k_res_add<<<gridFor(elemThreads), B256, 0, stream>>>(bufC, bufA, bres, NN);

  // L4: bufC -> bufB
  conv(bufC, Wh[2], b[3], bufB, HD);
  k_ln_relu<<<gridFor(waveThreads), B256, 0, stream>>>(bufB, gam[3], bet[3], NN);

  // L5: bufB -> bufC
  conv(bufB, Wh[3], b[4], bufC, HD);
  k_ln_relu<<<gridFor(waveThreads), B256, 0, stream>>>(bufC, gam[4], bet[4], NN);

  // L6: bufC -> bufB (no LN/ReLU)
  conv(bufC, Wh[4], b[5], bufB, HD);

  // ---- global mean pool + MLP head ----
  k_zero<<<gridFor(NG * HD), B256, 0, stream>>>(pooled, NG * HD);
  k_zero<<<1, B256, 0, stream>>>(cnt, NG);
  k_pool<<<gridFor(waveThreads), B256, 0, stream>>>(bufB, batch, pooled, cnt, NN);
  k_mlp<<<1, 64, 0, stream>>>(pooled, cnt, Wm1, bm1, Wm2, bm2, out);
}
